// Dinov3FlowHead_74028056313863
// MI455X (gfx1250) — matrix operand order repaired and checked by reading, not compiled, not run on hardware
//
#include <hip/hip_runtime.h>
#include <hip/hip_bf16.h>
#include <cstdint>

// ---------------- types ----------------
typedef __attribute__((ext_vector_type(16))) __bf16 v16bf;
typedef __attribute__((ext_vector_type(8)))  float  v8f;

// ---------------- constants ----------------
// B=32, C=384, H=W=40, RADIUS=4 -> 81 corr ch, HIDDEN=128
// x channels: [corr 0..80][f1 81..464][xx 465][yy 466][pad..479]
#define NB    32
#define NC    384
#define NH    40
#define NW    40
#define NPIX  (NB*NH*NW)     // 51200
#define CKIN  480            // padded 467 -> 480 (multiple of 32)
#define CINR  467
#define HID   128

__device__ __forceinline__ unsigned short f2bf(float f) {
  unsigned int u = __float_as_uint(f);
  unsigned int r = u + 0x7fffu + ((u >> 16) & 1u);
  return (unsigned short)(r >> 16);
}
__device__ __forceinline__ float bf2f(unsigned short h) {
  return __uint_as_float(((unsigned int)h) << 16);
}

// ---------------- zero fill ----------------
__global__ void fill_zero_kernel(uint4* __restrict__ p, long long n16) {
  long long i = (long long)blockIdx.x * blockDim.x + threadIdx.x;
  if (i < n16) { uint4 z = {0u,0u,0u,0u}; p[i] = z; }
}

// ---------------- BN fold: s = gamma*rsqrt(var+eps); t = beta + (b-mean)*s ----------------
struct BNIn5 {
  const float* g[5]; const float* be[5]; const float* mn[5];
  const float* vr[5]; const float* bb[5];
};
__global__ void bnprep_kernel(BNIn5 P, float* __restrict__ S, float* __restrict__ T) {
  int t = blockIdx.x * blockDim.x + threadIdx.x;
  if (t >= 5 * 128) return;
  int L = t >> 7, c = t & 127;
  float inv = rsqrtf(P.vr[L][c] + 1e-5f);
  float s = inv * P.g[L][c];
  S[t] = s;
  T[t] = P.be[L][c] + (P.bb[L][c] - P.mn[L][c]) * s;
}

// ---------------- weight repack into WMMA B-fragment order ----------------
// dst layout: [kt][nt 0..7][lane 0..31][e 0..15] bf16, B[k,n] fragment order:
//   klocal = (e<8 ? e : e+8) + (lane<16 ? 0 : 8), n = lane&15
// K ordering: k = (ky*3+kx)*CK + ci ; w src layout [co][ciR][ky][kx]
__global__ void repack_w_kernel(const float* __restrict__ w, int CinR, int CK, int KT,
                                unsigned short* __restrict__ dst) {
  int idx = blockIdx.x * 256 + threadIdx.x;
  int total = KT * 4096;
  if (idx >= total) return;
  int kt = idx / 4096; int r = idx & 4095;
  int nt = r >> 9;     int q = r & 511;
  int lane = q >> 4;   int e = q & 15;
  int klocal = ((e < 8) ? e : e + 8) + ((lane < 16) ? 0 : 8);
  int n = lane & 15;
  int k = kt * 32 + klocal;
  int plane = k / CK;  int ci = k - plane * CK;
  int ky = plane / 3,  kx = plane - ky * 3;
  int co = nt * 16 + n;
  float v = (ci < CinR) ? w[((co * CinR + ci) * 3 + ky) * 3 + kx] : 0.f;
  dst[idx] = f2bf(v);
}

// ---------------- pack f2 -> padded NHWC bf16 [B][48][48][384] + inv-norms ----------------
__global__ __launch_bounds__(256)
void pack_f2_kernel(const float* __restrict__ f2,
                    unsigned short* __restrict__ f2pad, float* __restrict__ n2pad) {
  __shared__ unsigned short lds[16 * 384];
  __shared__ float red[256];
  int t = threadIdx.x;
  int p0 = blockIdx.x * 16;
  {
    int pix = t & 15;
    int p = p0 + pix; int b = p / 1600; int rr = p - b * 1600;
    int y = rr / 40;  int x = rr - y * 40;
    float ss = 0.f;
    for (int c0 = 0; c0 < 384; c0 += 16) {
      int c = c0 + (t >> 4);
      float v = f2[(b * 384 + c) * 1600 + y * 40 + x];
      lds[pix * 384 + c] = f2bf(v);
      ss += v * v;
    }
    red[t] = ss;
  }
  __syncthreads();
  if (t < 16) {
    float s = 0.f;
    for (int g = 0; g < 16; ++g) s += red[(g << 4) + t];
    int p = p0 + t; int b = p / 1600; int rr = p - b * 1600;
    int y = rr / 40; int x = rr - y * 40;
    n2pad[(b * 48 + y + 4) * 48 + x + 4] = 1.0f / (sqrtf(s) + 1e-6f);
  }
  __syncthreads();
  {
    int pix = t >> 4, seg = t & 15;          // 24 ch = 48B = 3 uint4
    int p = p0 + pix; int b = p / 1600; int rr = p - b * 1600;
    int y = rr / 40;  int x = rr - y * 40;
    const uint4* src = (const uint4*)(lds + pix * 384 + seg * 24);
    uint4* dst = (uint4*)(f2pad + (((size_t)b * 48 + y + 4) * 48 + x + 4) * 384 + seg * 24);
    dst[0] = src[0]; dst[1] = src[1]; dst[2] = src[2];
  }
}

// ---------------- pack f1 -> f1nhwc bf16, x_pad channels 81..466, n1 ----------------
__global__ __launch_bounds__(256)
void pack_f1_kernel(const float* __restrict__ f1,
                    unsigned short* __restrict__ f1nhwc,
                    unsigned short* __restrict__ xpad,
                    float* __restrict__ n1) {
  __shared__ unsigned short lds[16 * 384];
  __shared__ float red[256];
  int t = threadIdx.x;
  int p0 = blockIdx.x * 16;
  {
    int pix = t & 15;
    int p = p0 + pix; int b = p / 1600; int rr = p - b * 1600;
    int y = rr / 40;  int x = rr - y * 40;
    float ss = 0.f;
    for (int c0 = 0; c0 < 384; c0 += 16) {
      int c = c0 + (t >> 4);
      float v = f1[(b * 384 + c) * 1600 + y * 40 + x];
      lds[pix * 384 + c] = f2bf(v);
      ss += v * v;
    }
    red[t] = ss;
  }
  __syncthreads();
  if (t < 16) {
    float s = 0.f;
    for (int g = 0; g < 16; ++g) s += red[(g << 4) + t];
    int p = p0 + t; int b = p / 1600; int rr = p - b * 1600;
    int y = rr / 40; int x = rr - y * 40;
    n1[(b * 40 + y) * 40 + x] = 1.0f / (sqrtf(s) + 1e-6f);
    unsigned short* xb = xpad + (((size_t)b * 42 + y + 1) * 42 + x + 1) * CKIN;
    xb[465] = f2bf((float)x);
    xb[466] = f2bf((float)y);
  }
  __syncthreads();
  {
    int pix = t >> 4, seg = t & 15;
    int p = p0 + pix; int b = p / 1600; int rr = p - b * 1600;
    int y = rr / 40;  int x = rr - y * 40;
    const uint4* src = (const uint4*)(lds + pix * 384 + seg * 24);
    uint4* dst = (uint4*)(f1nhwc + (((size_t)b * 40 + y) * 40 + x) * 384 + seg * 24);
    dst[0] = src[0]; dst[1] = src[1]; dst[2] = src[2];
    unsigned short* xb = xpad + (((size_t)b * 42 + y + 1) * 42 + x + 1) * CKIN + 81 + seg * 24;
    const unsigned short* ls = lds + pix * 384 + seg * 24;
    #pragma unroll
    for (int q = 0; q < 24; ++q) xb[q] = ls[q];
  }
}

// ---------------- local correlation: 8x8 pixel tile, 16x16x384 f2 tile in LDS ----------------
__global__ __launch_bounds__(256)
void corr_kernel(const unsigned short* __restrict__ f2pad,
                 const unsigned short* __restrict__ f1nhwc,
                 const float* __restrict__ n1,
                 const float* __restrict__ n2pad,
                 unsigned short* __restrict__ xpad) {
  extern __shared__ unsigned short sm[];
  unsigned short* f2s = sm;                 // [16][16][384] bf16 = 196608 B
  unsigned short* f1s = sm + 16 * 16 * 384; // [64][384]     bf16 =  49152 B
  int t = threadIdx.x;
  int blk = blockIdx.x;               // b*25 + ty*5 + tx
  int b = blk / 25; int r = blk - b * 25;
  int ty = r / 5;   int tx = r - ty * 5;
  int y0 = ty * 8, x0 = tx * 8;
  {
    int i = t >> 4, j = t & 15;       // one padded pixel row each
    const uint4* src = (const uint4*)(f2pad + (((size_t)b * 48 + y0 + i) * 48 + x0 + j) * 384);
    uint4* dst = (uint4*)(f2s + (i * 16 + j) * 384);
    #pragma unroll 8
    for (int q = 0; q < 48; ++q) dst[q] = src[q];
  }
  {
    int pix = t >> 2, seg = t & 3;    // 96 ch = 12 uint4
    int yy = y0 + (pix >> 3), xx = x0 + (pix & 7);
    const uint4* src = (const uint4*)(f1nhwc + (((size_t)b * 40 + yy) * 40 + xx) * 384 + seg * 96);
    uint4* dst = (uint4*)(f1s + pix * 384 + seg * 96);
    #pragma unroll
    for (int q = 0; q < 12; ++q) dst[q] = src[q];
  }
  __syncthreads();
  const float invs = 0.051031036307982884f;  // 1/sqrt(384)
  for (int task = t; task < 64 * 81; task += 256) {
    int pix = task / 81, o = task - pix * 81;
    int dy = o / 9 - 4, dx = o - (o / 9) * 9 - 4;
    int py = pix >> 3, px = pix & 7;
    const unsigned int* a = (const unsigned int*)(f1s + pix * 384);
    const unsigned int* bp = (const unsigned int*)(f2s + ((py + 4 + dy) * 16 + (px + 4 + dx)) * 384);
    float dot = 0.f;
    #pragma unroll 4
    for (int c = 0; c < 192; ++c) {
      unsigned int av = a[c], bv = bp[c];
      float a0 = __uint_as_float(av << 16), a1 = __uint_as_float(av & 0xffff0000u);
      float b0 = __uint_as_float(bv << 16), b1 = __uint_as_float(bv & 0xffff0000u);
      dot = fmaf(a0, b0, dot);
      dot = fmaf(a1, b1, dot);
    }
    int gy = y0 + py, gx = x0 + px;
    float v = dot * n1[(b * 40 + gy) * 40 + gx]
                  * n2pad[(b * 48 + gy + dy + 4) * 48 + gx + dx + 4] * invs;
    xpad[(((size_t)b * 42 + gy + 1) * 42 + gx + 1) * CKIN + o] = f2bf(v);
  }
}

// ---------------- implicit-GEMM 3x3 conv + BN + ReLU (+residual) via WMMA bf16 ----------------
// act: padded NHWC bf16 [B][42][42][CK]; out: padded NHWC bf16 [B][42][42][128]
// wave: M16 x N64 tile; block: 8 waves = 64 pixels x 128 Cout.
// Weight chunks (8 KB per K32 chunk) are streamed into LDS with CDNA5 async
// loads (GLOBAL_LOAD_ASYNC_TO_LDS_B128, ASYNCcnt), triple buffered so the copy
// of chunk kt+1 overlaps the WMMAs of chunk kt.
template<int CK, int KT>
__global__ __launch_bounds__(256)
void conv_wmma_kernel(const unsigned short* __restrict__ act,
                      const unsigned short* __restrict__ wfrag,
                      const float* __restrict__ bnS,
                      const float* __restrict__ bnT,
                      const unsigned short* __restrict__ resid,
                      unsigned short* __restrict__ out) {
  constexpr int CPP = CK / 32;   // K32 chunks per (ky,kx) plane
  __shared__ unsigned short wbuf[3][4096];   // 3 x 8 KB weight chunk buffers
  const int tid = threadIdx.x;
  const int lane = tid & 31;
  const int wave = tid >> 5;
  const int n0 = (wave & 1) * 64;
  const int m0 = blockIdx.x * 64 + (wave >> 1) * 16;

  // this lane's A-matrix pixel (M row = lane&15)
  const int p  = m0 + (lane & 15);
  const int pb = p / 1600;
  const int pr = p - pb * 1600;
  const int py = pr / 40;
  const int px = pr - py * 40;
  // 16-bit A layout: lanes<16 take bytes [0..15]+[32..47] of the 64B K-chunk, lanes>=16 the rest
  const unsigned short* abase =
      act + (((size_t)pb * 42 + py) * 42 + px) * CK + ((lane < 16) ? 0 : 8);

  // async weight-chunk copy: 256 threads x 32B = 8 KB per chunk
  const unsigned lbase = (unsigned)(size_t)(&wbuf[0][0]) + (unsigned)tid * 32u;
  const unsigned long long wsa = (unsigned long long)(size_t)wfrag;
  auto issue_chunk = [&](int kt, int buf) {
    unsigned voff  = (unsigned)kt * 8192u + (unsigned)tid * 32u;
    unsigned laddr = lbase + (unsigned)buf * 8192u;
    asm volatile("global_load_async_to_lds_b128 %0, %1, %2 offset:0\n\t"
                 "global_load_async_to_lds_b128 %3, %1, %2 offset:16"
                 :: "v"(laddr), "v"(voff), "s"(wsa), "v"(laddr + 16u)
                 : "memory");
  };

  v8f acc0 = {0.f,0.f,0.f,0.f,0.f,0.f,0.f,0.f};
  v8f acc1 = acc0, acc2 = acc0, acc3 = acc0;

  issue_chunk(0, 0);
  int kt = 0, cur = 0, nxt = 1;
  const unsigned short* wfragLDS = &wbuf[0][0] + ((n0 >> 4) * 512 + lane * 16);

  for (int ky = 0; ky < 3; ++ky) {
    for (int kx = 0; kx < 3; ++kx) {
      const unsigned short* ap0 = abase + (ky * 42 + kx) * CK;
      // prefetch next plane's activation row while this plane computes
      if (kx != 2 || ky != 2) {
        int nplane = ky * 3 + kx + 1;
        __builtin_prefetch(abase + ((nplane / 3) * 42 + (nplane % 3)) * CK, 0, 0);
      }
      #pragma unroll
      for (int c = 0; c < CPP; ++c, ++kt) {
        if (kt + 1 < KT) issue_chunk(kt + 1, nxt);
        // two just-issued copies may be outstanding; older ones must be done
        asm volatile("s_wait_asynccnt 0x2" ::: "memory");
        __syncthreads();

        union FU { uint4 q[2]; v16bf v; } A, B0, B1, B2, B3;
        const unsigned short* ap = ap0 + c * 32;
        A.q[0] = *(const uint4*)(ap);
        A.q[1] = *(const uint4*)(ap + 16);

        const unsigned short* wp = wfragLDS + cur * 4096;
        B0.q[0] = *(const uint4*)(wp);         B0.q[1] = *(const uint4*)(wp + 8);
        B1.q[0] = *(const uint4*)(wp + 512);   B1.q[1] = *(const uint4*)(wp + 520);
        B2.q[0] = *(const uint4*)(wp + 1024);  B2.q[1] = *(const uint4*)(wp + 1032);
        B3.q[0] = *(const uint4*)(wp + 1536);  B3.q[1] = *(const uint4*)(wp + 1544);

        acc0 = __builtin_amdgcn_wmma_f32_16x16x32_bf16(false, A.v, false, B0.v, (short)0, acc0, false, false);
        acc1 = __builtin_amdgcn_wmma_f32_16x16x32_bf16(false, A.v, false, B1.v, (short)0, acc1, false, false);
        acc2 = __builtin_amdgcn_wmma_f32_16x16x32_bf16(false, A.v, false, B2.v, (short)0, acc2, false, false);
        acc3 = __builtin_amdgcn_wmma_f32_16x16x32_bf16(false, A.v, false, B3.v, (short)0, acc3, false, false);

        cur = nxt;
        nxt = (nxt == 2) ? 0 : nxt + 1;
      }
    }
  }

  // C layout: VGPR i, lanes 0-15 -> M=i, N=lane; lanes 16-31 -> M=8+i, N=lane-16
  const int mbase = (lane < 16) ? 0 : 8;
  auto store_tile = [&](v8f a, int nt) {
    const int co = n0 + nt * 16 + (lane & 15);
    const float s = bnS[co];
    const float tt = bnT[co];
    #pragma unroll
    for (int i = 0; i < 8; ++i) {
      const int pp = m0 + mbase + i;
      const int bb = pp / 1600;
      const int rr = pp - bb * 1600;
      const int yy = rr / 40;
      const int xx = rr - yy * 40;
      const size_t oidx = (((size_t)bb * 42 + yy + 1) * 42 + xx + 1) * 128 + co;
      float v = fmaf(a[i], s, tt);
      v = v > 0.f ? v : 0.f;
      if (resid) v += bf2f(resid[oidx]);
      out[oidx] = f2bf(v);
    }
  };
  store_tile(acc0, 0); store_tile(acc1, 1); store_tile(acc2, 2); store_tile(acc3, 3);
}

// ---------------- prediction conv 128 -> 2, fp32 accumulate, NCHW output ----------------
__global__ __launch_bounds__(256)
void pred_kernel(const unsigned short* __restrict__ xf,
                 const float* __restrict__ pw, const float* __restrict__ pb,
                 float* __restrict__ out) {
  int t = blockIdx.x * 256 + threadIdx.x;   // over 32*2*1600
  if (t >= NB * 2 * 1600) return;
  int b = t / 3200; int r = t - b * 3200;
  int co = r / 1600; int r2 = r - co * 1600;
  int y = r2 / 40; int x = r2 - y * 40;
  float acc = pb[co];
  for (int ky = 0; ky < 3; ++ky)
    for (int kx = 0; kx < 3; ++kx) {
      const unsigned short* ap = xf + (((size_t)b * 42 + y + ky) * 42 + x + kx) * 128;
      const float* wp = pw + (size_t)co * 128 * 9 + ky * 3 + kx;  // w[co][ci][ky][kx]
      #pragma unroll 4
      for (int ci = 0; ci < 128; ++ci)
        acc = fmaf(bf2f(ap[ci]), wp[ci * 9], acc);
    }
  out[t] = acc;
}

// ---------------- bilinear upsample 40x40 -> 640x640, x16 scale ----------------
__global__ __launch_bounds__(256)
void upsample_kernel(const float* __restrict__ ff, float* __restrict__ out) {
  int idx = blockIdx.x * 256 + threadIdx.x;   // over 32*2*640*160 (4 cols per thread)
  if (idx >= NB * 2 * 640 * 160) return;
  int jb = idx % 160; int r = idx / 160;
  int i = r % 640;    int r2 = r / 640;       // r2 = b*2 + co
  const float step = 39.0f / 639.0f;
  float sy = i * step;
  int i0 = (int)sy; if (i0 > 38) i0 = 38;
  float tv = sy - (float)i0;
  const float* row0 = ff + (size_t)r2 * 1600 + i0 * 40;
  const float* row1 = row0 + 40;
  float4 res;
  #pragma unroll
  for (int u = 0; u < 4; ++u) {
    int j = jb * 4 + u;
    float sx = j * step;
    int j0 = (int)sx; if (j0 > 38) j0 = 38;
    float tu = sx - (float)j0;
    float a = row0[j0] + (row0[j0 + 1] - row0[j0]) * tu;
    float c = row1[j0] + (row1[j0 + 1] - row1[j0]) * tu;
    float v = a + (c - a) * tv;
    (&res.x)[u] = v * 16.0f;
  }
  *(float4*)(out + (size_t)NB * 2 * 1600 + (size_t)idx * 4) = res;
}

// ---------------- host launch ----------------
struct ConvP { const float *w, *b, *g, *be, *mn, *vr; };

extern "C" void kernel_launch(void* const* d_in, const int* in_sizes, int n_in,
                              void* d_out, int out_size, void* d_ws, size_t ws_size,
                              hipStream_t stream) {
  const float* f1 = (const float*)d_in[0];
  const float* f2 = (const float*)d_in[1];
  ConvP L[5];     // 0..2 stem, 3..4 refine
  const float* predw = nullptr;
  const float* predb = nullptr;
  if (n_in >= 4 && in_sizes[2] == 2) {
    // JAX pytree (sorted keys): pred_b, pred_w, refine[0..1]{b,beta,gamma,mean,var,w}, stem[0..2]{...}
    predb = (const float*)d_in[2];
    predw = (const float*)d_in[3];
    int i = 4;
    for (int r = 0; r < 2; ++r) {
      ConvP& c = L[3 + r];
      c.b  = (const float*)d_in[i];     c.be = (const float*)d_in[i + 1];
      c.g  = (const float*)d_in[i + 2]; c.mn = (const float*)d_in[i + 3];
      c.vr = (const float*)d_in[i + 4]; c.w  = (const float*)d_in[i + 5];
      i += 6;
    }
    for (int s = 0; s < 3; ++s) {
      ConvP& c = L[s];
      c.b  = (const float*)d_in[i];     c.be = (const float*)d_in[i + 1];
      c.g  = (const float*)d_in[i + 2]; c.mn = (const float*)d_in[i + 3];
      c.vr = (const float*)d_in[i + 4]; c.w  = (const float*)d_in[i + 5];
      i += 6;
    }
  } else {
    // insertion order: stem[0..2]{w,b,gamma,beta,mean,var}, refine[0..1]{...}, pred_w, pred_b
    int i = 2;
    for (int s = 0; s < 5; ++s) {
      ConvP& c = L[s];
      c.w  = (const float*)d_in[i];     c.b  = (const float*)d_in[i + 1];
      c.g  = (const float*)d_in[i + 2]; c.be = (const float*)d_in[i + 3];
      c.mn = (const float*)d_in[i + 4]; c.vr = (const float*)d_in[i + 5];
      i += 6;
    }
    predw = (const float*)d_in[i];
    predb = (const float*)d_in[i + 1];
  }

  // ---- carve workspace ----
  char* base = (char*)d_ws;
  size_t off = 0;
  auto carve = [&](size_t bytes) -> void* {
    void* pp = (void*)(base + off);
    off = (off + bytes + 255) & ~(size_t)255;
    return pp;
  };
  unsigned short* f2pad = (unsigned short*)carve((size_t)NB * 48 * 48 * 384 * 2);
  unsigned short* xpad  = (unsigned short*)carve((size_t)NB * 42 * 42 * CKIN * 2);
  unsigned short* h1    = (unsigned short*)carve((size_t)NB * 42 * 42 * 128 * 2);
  unsigned short* h2    = (unsigned short*)carve((size_t)NB * 42 * 42 * 128 * 2);
  unsigned short* h3    = (unsigned short*)carve((size_t)NB * 42 * 42 * 128 * 2);
  unsigned short* rr1   = (unsigned short*)carve((size_t)NB * 42 * 42 * 128 * 2);
  unsigned short* xfin  = (unsigned short*)carve((size_t)NB * 42 * 42 * 128 * 2);
  float* n2pad          = (float*)carve((size_t)NB * 48 * 48 * 4);
  size_t zend = off;    // everything up to here is zero-filled (halos need it)
  unsigned short* f1nhwc = (unsigned short*)carve((size_t)NB * 1600 * 384 * 2);
  float* n1              = (float*)carve((size_t)NB * 1600 * 4);
  unsigned short* wf0    = (unsigned short*)carve((size_t)135 * 4096 * 2);
  unsigned short* wfh[4];
  for (int k = 0; k < 4; ++k) wfh[k] = (unsigned short*)carve((size_t)36 * 4096 * 2);
  float* bnS = (float*)carve(640 * 4);
  float* bnT = (float*)carve(640 * 4);
  (void)ws_size; (void)out_size;

  // ---- zero halo'd buffers ----
  long long n16 = (long long)(zend / 16);
  fill_zero_kernel<<<(unsigned)((n16 + 255) / 256), 256, 0, stream>>>((uint4*)base, n16);

  // ---- BN fold + weight repack ----
  BNIn5 bn;
  for (int l = 0; l < 5; ++l) {
    bn.g[l] = L[l].g; bn.be[l] = L[l].be; bn.mn[l] = L[l].mn;
    bn.vr[l] = L[l].vr; bn.bb[l] = L[l].b;
  }
  bnprep_kernel<<<3, 256, 0, stream>>>(bn, bnS, bnT);
  repack_w_kernel<<<(135 * 4096 + 255) / 256, 256, 0, stream>>>(L[0].w, CINR, CKIN, 135, wf0);
  for (int l = 0; l < 4; ++l)
    repack_w_kernel<<<(36 * 4096 + 255) / 256, 256, 0, stream>>>(L[1 + l].w, 128, 128, 36, wfh[l]);

  // ---- pack features ----
  pack_f2_kernel<<<NPIX / 16, 256, 0, stream>>>(f2, f2pad, n2pad);
  pack_f1_kernel<<<NPIX / 16, 256, 0, stream>>>(f1, f1nhwc, xpad, n1);

  // ---- correlation (245760 B dynamic LDS; WGP has 320 KB) ----
  hipFuncSetAttribute((const void*)corr_kernel,
                      hipFuncAttributeMaxDynamicSharedMemorySize, 245760);
  corr_kernel<<<NB * 25, 256, 245760, stream>>>(f2pad, f1nhwc, n1, n2pad, xpad);

  // ---- conv stack (implicit GEMM, bf16 WMMA + async weight streaming) ----
  const unsigned GB = NPIX / 64;  // 800 blocks
  conv_wmma_kernel<CKIN, 135><<<GB, 256, 0, stream>>>(xpad, wf0, bnS +   0, bnT +   0, nullptr, h1);
  conv_wmma_kernel<128,  36 ><<<GB, 256, 0, stream>>>(h1,  wfh[0], bnS + 128, bnT + 128, nullptr, h2);
  conv_wmma_kernel<128,  36 ><<<GB, 256, 0, stream>>>(h2,  wfh[1], bnS + 256, bnT + 256, nullptr, h3);
  conv_wmma_kernel<128,  36 ><<<GB, 256, 0, stream>>>(h3,  wfh[2], bnS + 384, bnT + 384, nullptr, rr1);
  conv_wmma_kernel<128,  36 ><<<GB, 256, 0, stream>>>(rr1, wfh[3], bnS + 512, bnT + 512, h3,      xfin);

  // ---- prediction + upsample ----
  float* out = (float*)d_out;
  pred_kernel<<<(NB * 2 * 1600 + 255) / 256, 256, 0, stream>>>(xfin, predw, predb, out);
  upsample_kernel<<<(NB * 2 * 640 * 160 + 255) / 256, 256, 0, stream>>>(out, out);
}
